// GAT_87952340287704
// MI455X (gfx1250) — compile-verified, matrix operand order checked by reading, not run on hardware
//
#include <hip/hip_runtime.h>
#include <hip/hip_bf16.h>

typedef __bf16 bf16;
typedef __attribute__((ext_vector_type(16))) __bf16 v16bf;
typedef __attribute__((ext_vector_type(8)))  float  v8f;

// gcc-style vectors to match gfx1250 builtin signatures
typedef unsigned int u32x4 __attribute__((vector_size(16)));
typedef int          i32x4 __attribute__((vector_size(16)));
typedef int          i32x8 __attribute__((vector_size(32)));

// address-space-qualified VECTOR types (qualifier on the vector, not the element)
typedef int ivec4 __attribute__((vector_size(16)));
typedef __attribute__((address_space(1))) ivec4 gvec4;
typedef __attribute__((address_space(3))) ivec4 lvec4;

#define U_N       50000
#define T_N       2048
#define D_F       128
#define H_N       4
#define M_REPO_N  4096
#define E_TEAM_N  32768
#define E_USER_T  850000   // 800000 random + 50000 self-loops baked into edge list
#define NMAX      52096    // >= max padded n (52048 -> 52096)

// --- feature detection for CDNA5 async paths (fallback keeps build green) ---
#if defined(__has_builtin)
#if __has_builtin(__builtin_amdgcn_global_load_async_to_lds_b128)
#define HAS_ASYNC 1
#endif
#if __has_builtin(__builtin_amdgcn_tensor_load_to_lds) && __has_builtin(__builtin_amdgcn_s_wait_tensorcnt)
#define HAS_TDM 1
#endif
#endif

#if defined(HAS_ASYNC)
#if __has_builtin(__builtin_amdgcn_s_wait_asynccnt)
#define WAIT_ASYNC0() __builtin_amdgcn_s_wait_asynccnt(0)
#else
#define WAIT_ASYNC0() asm volatile("s_wait_asynccnt 0x0" ::: "memory")
#endif
#endif

typedef __attribute__((address_space(1))) bf16  gbf16;
typedef __attribute__((address_space(3))) bf16  lbf16;

// ---------------------------------------------------------------------------
// f32 -> bf16 elementwise convert
__global__ void cvt_bf16_kernel(bf16* __restrict__ dst, const float* __restrict__ src, size_t n) {
    size_t i = (size_t)blockIdx.x * 256 + threadIdx.x;
    if (i < n) dst[i] = (bf16)src[i];
}

// Transpose + convert weights: dst[ncol*K + k] = (bf16) src[k*N + ncol]   (B^T, row = out col)
__global__ void wtrans_kernel(bf16* __restrict__ dst, const float* __restrict__ src, int K, int N) {
    int t = blockIdx.x * 256 + threadIdx.x;
    if (t >= K * N) return;
    int ncol = t / K, k = t % K;
    dst[t] = (bf16)src[(size_t)k * N + ncol];
}

// ---------------------------------------------------------------------------
// C[npad x 128] = A[npad x K] * B[K x 128], B supplied transposed: BT[128 x K].
// Block = 256 threads (8 waves), block tile M=128 x N=128.
//  * B: one 128x128 bf16 K-chunk DMA'd to LDS by wave 0 via TDM (tensorcnt),
//       or cooperative async loads as fallback.
//  * A: 128x32 bf16 tiles, double-buffered via GLOBAL_LOAD_ASYNC_TO_LDS_B128
//       (asynccnt), issued for tile i+1 while WMMAs run on tile i.
// Dynamic LDS: 128*128*2 (Bs) + 2*128*32*2 (As ping/pong) = 48 KB.
__global__ __launch_bounds__(256)
void gemm_bf16_kernel(float* __restrict__ C, const bf16* __restrict__ A,
                      const bf16* __restrict__ BT, int K) {
    extern __shared__ bf16 smem[];

    const int  tid  = threadIdx.x;
    const int  wave = tid >> 5;
    const int  lane = tid & 31;
    const long rowBase = (long)blockIdx.x * 128;

    v8f acc[8];
#pragma unroll
    for (int t = 0; t < 8; ++t) acc[t] = (v8f){0.f, 0.f, 0.f, 0.f, 0.f, 0.f, 0.f, 0.f};

    const int half = (lane < 16) ? 0 : 1;   // K sub-block select (ISA 16-bit A/B layout)
    const int mrow = lane & 15;
    const int lo   = half * 8;              // bf16 element offset of first 16B chunk

#if defined(HAS_ASYNC)
    bf16* Bs  = smem;                 // 128 rows (out cols) x 128 k, stride 128
    bf16* As0 = smem + 128 * 128;     // 128 rows x 32 k, stride 32
    bf16* As1 = As0 + 128 * 32;

    gbf16* gA  = (gbf16*)A;
    gbf16* gBT = (gbf16*)BT;

    // per-thread A staging: 2 chunks of 8 bf16 (32B contiguous per thread)
    const int ca = tid * 2;
    const int rA = ca >> 2;            // 0..127
    const int kA = (ca & 3) << 3;      // 0, 8, 16, 24 (two consecutive)

    auto issueA = [&](int idx, bf16* buf) {   // idx = kb / 32
        const int kb = idx << 5;
        lbf16* l = (lbf16*)(buf + rA * 32 + kA);
        gbf16* g = gA + (rowBase + rA) * K + kb + kA;
        __builtin_amdgcn_global_load_async_to_lds_b128((gvec4*)g,       (lvec4*)l,       0, 0);
        __builtin_amdgcn_global_load_async_to_lds_b128((gvec4*)(g + 8), (lvec4*)(l + 8), 0, 0);
    };

#if defined(HAS_TDM)
    // TDM descriptor for one 128(out-col) x 128(k) bf16 tile of BT, row stride K.
    const unsigned bsOff = (unsigned)(unsigned long long)(lbf16*)Bs;
    auto issueB = [&](int c) {                 // c = K-chunk index
        unsigned long long ga = (unsigned long long)(gbf16*)(gBT + (long)c * 128);
        u32x4 g0;
        g0[0] = 1u;                              // count=1, user mode
        g0[1] = bsOff;                           // lds_addr
        g0[2] = (unsigned)ga;                    // global_addr[31:0]
        g0[3] = (unsigned)((ga >> 32) & 0x1FFFFFFull) | (2u << 30);  // addr[56:32] | type=2
        i32x8 g1;
        g1[0] = (1 << 16);                       // data_size = 1 (2 bytes)
        g1[1] = (128 << 16);                     // tensor_dim0[15:0] = 128
        g1[2] = (128 << 16);                     // tensor_dim0[31:16]=0 | tensor_dim1[15:0]=128
        g1[3] = (128 << 16);                     // tensor_dim1[31:16]=0 | tile_dim0=128
        g1[4] = 128;                             // tile_dim1=128, tile_dim2=0
        g1[5] = K;                               // tensor_dim0_stride[31:0] = K elements
        g1[6] = 0;                               // stride0[47:32]=0 | stride1[15:0]=0
        g1[7] = 0;                               // stride1[47:16]=0
        i32x4 z4 = (i32x4){0, 0, 0, 0};
        i32x8 z8 = (i32x8){0, 0, 0, 0, 0, 0, 0, 0};
        __builtin_amdgcn_tensor_load_to_lds(g0, g1, z4, z4, z8, 0);
    };
#else
    auto issueB = [&](int c) {                 // cooperative async loads: 2048 16B chunks
#pragma unroll
        for (int i = 0; i < 8; ++i) {
            int ch = tid + i * 256;            // 0..2047
            int r  = ch >> 4;                  // 0..127
            int kc = (ch & 15) << 3;           // 0..120
            __builtin_amdgcn_global_load_async_to_lds_b128(
                (gvec4*)(gBT + (long)r * K + c * 128 + kc),
                (lvec4*)((lbf16*)(Bs + r * 128 + kc)), 0, 0);
        }
    };
#endif

    const int nSteps = K >> 5;                 // 4 or 16
    const int nChunk = K >> 7;                 // 1 or 4

    issueA(0, As0);
#if defined(HAS_TDM)
    if (wave == 0) issueB(0);
#else
    issueB(0);
#endif

    for (int c = 0; c < nChunk; ++c) {
#if defined(HAS_TDM)
        if (wave == 0) __builtin_amdgcn_s_wait_tensorcnt(0);
        __syncthreads();                       // B chunk c visible to all waves
#endif
#pragma unroll
        for (int j = 0; j < 4; ++j) {
            const int idx = c * 4 + j;
            WAIT_ASYNC0();                     // A tile idx (and non-TDM B chunk) arrived
            __syncthreads();
            if (idx + 1 < nSteps) issueA(idx + 1, (idx & 1) ? As0 : As1);
            const bf16* Ac = (idx & 1) ? As1 : As0;

            v16bf afr;
            {
                bf16* p = (bf16*)&afr;
                const int r = wave * 16 + mrow;
                *(float4*)(p)     = *(const float4*)(&Ac[r * 32 + lo]);
                *(float4*)(p + 8) = *(const float4*)(&Ac[r * 32 + lo + 16]);
            }
#pragma unroll
            for (int t = 0; t < 8; ++t) {
                v16bf bfr;
                bf16* p = (bf16*)&bfr;
                const int nn = t * 16 + mrow;
                *(float4*)(p)     = *(const float4*)(&Bs[nn * 128 + j * 32 + lo]);
                *(float4*)(p + 8) = *(const float4*)(&Bs[nn * 128 + j * 32 + lo + 16]);
                acc[t] = __builtin_amdgcn_wmma_f32_16x16x32_bf16(
                    false, afr, false, bfr, (short)0, acc[t], false, false);
            }
        }
        __syncthreads();                       // all waves done reading B chunk c
        if (c + 1 < nChunk) {
#if defined(HAS_TDM)
            if (wave == 0) issueB(c + 1);
#else
            issueB(c + 1);
#endif
        }
    }
#else  // ------------------- fully synchronous fallback ----------------------
    bf16* As = smem;                  // 128 x 32
    bf16* Bs = smem + 128 * 32;       // 128 x 32

    for (int kb = 0; kb < K; kb += 32) {
#pragma unroll
        for (int i = 0; i < 2; ++i) {
            int cc = tid + i * 256;
            int r  = cc >> 2;
            int kc = (cc & 3) << 3;
            *(float4*)(&As[r * 32 + kc]) = *(const float4*)(&A[(rowBase + r) * K + kb + kc]);
            *(float4*)(&Bs[r * 32 + kc]) = *(const float4*)(&BT[(long)r * K + kb + kc]);
        }
        __syncthreads();
        v16bf afr;
        {
            bf16* p = (bf16*)&afr;
            const int r = wave * 16 + mrow;
            *(float4*)(p)     = *(const float4*)(&As[r * 32 + lo]);
            *(float4*)(p + 8) = *(const float4*)(&As[r * 32 + lo + 16]);
        }
#pragma unroll
        for (int t = 0; t < 8; ++t) {
            v16bf bfr;
            bf16* p = (bf16*)&bfr;
            const int nn = t * 16 + mrow;
            *(float4*)(p)     = *(const float4*)(&Bs[nn * 32 + lo]);
            *(float4*)(p + 8) = *(const float4*)(&Bs[nn * 32 + lo + 16]);
            acc[t] = __builtin_amdgcn_wmma_f32_16x16x32_bf16(
                false, afr, false, bfr, (short)0, acc[t], false, false);
        }
        __syncthreads();
    }
#endif

    // ---- writeback: C/D layout -> lane<16: M=r, lane>=16: M=r+8, N = tile*16 + (lane&15)
#pragma unroll
    for (int t = 0; t < 8; ++t) {
        const int col = t * 16 + mrow;
#pragma unroll
        for (int r = 0; r < 8; ++r) {
            const long row = rowBase + wave * 16 + r + half * 8;
            C[row * 128 + col] = acc[t][r];
        }
    }
}

// ---------------------------------------------------------------------------
// Per-node attention scores: SS[i] = h[i]·a[0:128], SD[i] = h[i]·a[128:256]
__global__ void score_kernel(float* __restrict__ SS, float* __restrict__ SD,
                             const float* __restrict__ HT, const float* __restrict__ a, int n) {
    int i = blockIdx.x * 256 + threadIdx.x;
    if (i >= n) return;
    const float4* h  = (const float4*)(HT + (size_t)i * 128);
    const float4* a0 = (const float4*)(a);
    const float4* a1 = (const float4*)(a + 128);
    float ss = 0.f, sd = 0.f;
#pragma unroll 8
    for (int j = 0; j < 32; ++j) {
        float4 hv = h[j], av = a0[j], bv = a1[j];
        ss += hv.x * av.x + hv.y * av.y + hv.z * av.z + hv.w * av.w;
        sd += hv.x * bv.x + hv.y * bv.y + hv.z * bv.z + hv.w * bv.w;
    }
    SS[i] = ss;
    SD[i] = sd;
}

// Self-loop contribution, plain stores (runs first): one wave per row.
__global__ void selfloop_kernel(float* __restrict__ RS, float* __restrict__ NUM,
                                const float* __restrict__ SS, const float* __restrict__ SD,
                                const float* __restrict__ HT, int n) {
    int i = blockIdx.x * 8 + (threadIdx.x >> 5);
    if (i >= n) return;
    int lane = threadIdx.x & 31;
    float sc = SS[i] + SD[i];
    float lr = sc > 0.f ? sc : 0.2f * sc;
    float w  = __expf(-lr);
    if (lane == 0) RS[i] = w;
    float4 hv = ((const float4*)(HT + (size_t)i * 128))[lane];
    float4 o;
    o.x = w * hv.x; o.y = w * hv.y; o.z = w * hv.z; o.w = w * hv.w;
    ((float4*)(NUM + (size_t)i * 128))[lane] = o;
}

// Edge scatter: one wave per edge, lane handles 4 features. Atomic accumulate.
__global__ void edge_kernel(const int* __restrict__ src_idx, const int* __restrict__ dst_idx,
                            int nE, int src_add, int dst_const /* <0 => from dst_idx */,
                            const float* __restrict__ SS, const float* __restrict__ SD,
                            const float* __restrict__ HT,
                            float* __restrict__ RS, float* __restrict__ NUM) {
    int e = blockIdx.x * 8 + (threadIdx.x >> 5);
    if (e >= nE) return;
    int lane = threadIdx.x & 31;
    int s = src_idx[e] + src_add;
    int d = (dst_const >= 0) ? dst_const : dst_idx[e];
    float sc = SS[s] + SD[d];
    float lr = sc > 0.f ? sc : 0.2f * sc;
    float w  = __expf(-lr);
    if (lane == 0) atomicAdd(&RS[s], w);
    float4 hv = ((const float4*)(HT + (size_t)d * 128))[lane];
    float* np = NUM + (size_t)s * 128 + lane * 4;
    atomicAdd(np + 0, w * hv.x);
    atomicAdd(np + 1, w * hv.y);
    atomicAdd(np + 2, w * hv.z);
    atomicAdd(np + 3, w * hv.w);
}

// elu(num/rowsum) -> bf16 dest (into [n,512] xh at a column offset)
__global__ void finalize_bf16_kernel(bf16* __restrict__ dst, int ldd,
                                     const float* __restrict__ NUM, const float* __restrict__ RS, int n) {
    int i = blockIdx.x * 8 + (threadIdx.x >> 5);
    if (i >= n) return;
    int lane = threadIdx.x & 31;
    float r  = RS[i];
    float4 v = ((const float4*)(NUM + (size_t)i * 128))[lane];
    float o0 = v.x / r, o1 = v.y / r, o2 = v.z / r, o3 = v.w / r;
    o0 = o0 > 0.f ? o0 : __expf(o0) - 1.f;
    o1 = o1 > 0.f ? o1 : __expf(o1) - 1.f;
    o2 = o2 > 0.f ? o2 : __expf(o2) - 1.f;
    o3 = o3 > 0.f ? o3 : __expf(o3) - 1.f;
    bf16* p = dst + (size_t)i * ldd + lane * 4;
    p[0] = (bf16)o0; p[1] = (bf16)o1; p[2] = (bf16)o2; p[3] = (bf16)o3;
}

// elu(num/rowsum) -> f32 dest (block output)
__global__ void finalize_f32_kernel(float* __restrict__ dst, int ldd,
                                    const float* __restrict__ NUM, const float* __restrict__ RS, int n) {
    int i = blockIdx.x * 8 + (threadIdx.x >> 5);
    if (i >= n) return;
    int lane = threadIdx.x & 31;
    float r  = RS[i];
    float4 v = ((const float4*)(NUM + (size_t)i * 128))[lane];
    float4 o;
    o.x = v.x / r; o.y = v.y / r; o.z = v.z / r; o.w = v.w / r;
    o.x = o.x > 0.f ? o.x : __expf(o.x) - 1.f;
    o.y = o.y > 0.f ? o.y : __expf(o.y) - 1.f;
    o.z = o.z > 0.f ? o.z : __expf(o.z) - 1.f;
    o.w = o.w > 0.f ? o.w : __expf(o.w) - 1.f;
    ((float4*)(dst + (size_t)i * ldd))[lane] = o;
}

// sigmoid(team_h @ out_W^T + out_b)
__global__ void head_kernel(float* __restrict__ out, const float* __restrict__ YU,
                            const float* __restrict__ oW, const float* __restrict__ ob, int T) {
    int t = blockIdx.x * 256 + threadIdx.x;
    if (t >= T) return;
    const float4* h = (const float4*)(YU + (size_t)t * 128);
    const float4* w = (const float4*)oW;
    float s = 0.f;
#pragma unroll 8
    for (int j = 0; j < 32; ++j) {
        float4 hv = h[j], wv = w[j];
        s += hv.x * wv.x + hv.y * wv.y + hv.z * wv.z + hv.w * wv.w;
    }
    s += ob[0];
    out[t] = 1.f / (1.f + __expf(-s));
}

// ---------------------------------------------------------------------------
extern "C" void kernel_launch(void* const* d_in, const int* in_sizes, int n_in,
                              void* d_out, int out_size, void* d_ws, size_t ws_size,
                              hipStream_t stream) {
    const float* repo       = (const float*)d_in[0];
    const float* users      = (const float*)d_in[1];
    const float* teams      = (const float*)d_in[2];
    const int*   repo_users = (const int*)d_in[3];
    const int*   tu_team    = (const int*)d_in[4];
    const int*   tu_user    = (const int*)d_in[5];
    const int*   user_edges = (const int*)d_in[6];
    const float* W_in[3]    = { (const float*)d_in[7],  (const float*)d_in[11], (const float*)d_in[15] };
    const float* a_in[3]    = { (const float*)d_in[8],  (const float*)d_in[12], (const float*)d_in[16] };
    const float* oW_in[3]   = { (const float*)d_in[9],  (const float*)d_in[13], (const float*)d_in[17] };
    const float* oa_in[3]   = { (const float*)d_in[10], (const float*)d_in[14], (const float*)d_in[18] };
    const float* out_W      = (const float*)d_in[19];
    const float* out_b      = (const float*)d_in[20];
    float* out              = (float*)d_out;

#if defined(HAS_ASYNC)
    const size_t GEMM_SH = (size_t)(128 * 128 + 2 * 128 * 32) * sizeof(bf16);  // 48 KB
#else
    const size_t GEMM_SH = (size_t)(2 * 128 * 32) * sizeof(bf16);              // 16 KB
#endif

    // workspace carve-up
    char*  ws   = (char*)d_ws;
    auto   take = [&](size_t bytes) { char* p = ws; ws += (bytes + 255) & ~(size_t)255; return p; };
    bf16*  XB128 = (bf16*)take((size_t)NMAX * 128 * sizeof(bf16));
    bf16*  XB512 = (bf16*)take((size_t)NMAX * 512 * sizeof(bf16));
    float* HT    = (float*)take((size_t)NMAX * 128 * sizeof(float));
    float* NUM   = (float*)take((size_t)NMAX * 128 * sizeof(float));
    float* YOUT  = (float*)take((size_t)NMAX * 128 * sizeof(float));
    float* SS    = (float*)take((size_t)NMAX * sizeof(float));
    float* SD    = (float*)take((size_t)NMAX * sizeof(float));
    float* RS    = (float*)take((size_t)NMAX * sizeof(float));
    bf16*  WT    = (bf16*)take((size_t)H_N * 128 * 128 * sizeof(bf16));
    bf16*  OWT   = (bf16*)take((size_t)128 * 512 * sizeof(bf16));

    auto cvt = [&](bf16* dst, const float* src, size_t cnt) {
        cvt_bf16_kernel<<<dim3((unsigned)((cnt + 255) / 256)), 256, 0, stream>>>(dst, src, cnt);
    };

    // One full gat_block (4 heads + out layer)
    auto gat_block = [&](int n, int npad,
                         const float* W, const float* a, const float* oW, const float* oa,
                         const int* esrc, const int* edst, int nE,
                         int src_add, int dst_const, bool do_selfloop) {
        // weights -> bf16, transposed
        for (int h = 0; h < H_N; ++h)
            wtrans_kernel<<<dim3((128 * 128 + 255) / 256), 256, 0, stream>>>(
                WT + (size_t)h * 128 * 128, W + (size_t)h * 128 * 128, 128, 128);
        wtrans_kernel<<<dim3((512 * 128 + 255) / 256), 256, 0, stream>>>(OWT, oW, 512, 128);
        (void)hipMemsetAsync(XB512, 0, (size_t)NMAX * 512 * sizeof(bf16), stream);

        auto one_layer = [&](const bf16* Abuf, const bf16* BTbuf, int K, const float* avec,
                             bool to_bf16, int col_ofs) {
            gemm_bf16_kernel<<<dim3(npad / 128), 256, GEMM_SH, stream>>>(HT, Abuf, BTbuf, K);
            score_kernel<<<dim3((n + 255) / 256), 256, 0, stream>>>(SS, SD, HT, avec, n);
            if (do_selfloop) {
                selfloop_kernel<<<dim3((n + 7) / 8), 256, 0, stream>>>(RS, NUM, SS, SD, HT, n);
            } else {
                (void)hipMemsetAsync(RS, 0, (size_t)npad * sizeof(float), stream);
                (void)hipMemsetAsync(NUM, 0, (size_t)npad * 128 * sizeof(float), stream);
            }
            edge_kernel<<<dim3((nE + 7) / 8), 256, 0, stream>>>(
                esrc, edst, nE, src_add, dst_const, SS, SD, HT, RS, NUM);
            if (to_bf16)
                finalize_bf16_kernel<<<dim3((n + 7) / 8), 256, 0, stream>>>(
                    XB512 + col_ofs, 512, NUM, RS, n);
            else
                finalize_f32_kernel<<<dim3((n + 7) / 8), 256, 0, stream>>>(YOUT, 128, NUM, RS, n);
        };

        for (int h = 0; h < H_N; ++h)
            one_layer(XB128, WT + (size_t)h * 128 * 128, 128, a + (size_t)h * 256, true, h * 128);
        one_layer(XB512, OWT, 512, oa, false, 0);
    };

    // ---------------- phase 1: repo (nodes = users + repo node) ----------------
    {
        const int n = U_N + 1, npad = ((n + 127) / 128) * 128;
        (void)hipMemsetAsync(XB128, 0, (size_t)NMAX * 128 * sizeof(bf16), stream);
        cvt(XB128, users, (size_t)U_N * 128);
        cvt(XB128 + (size_t)U_N * 128, repo, 128);
        gat_block(n, npad, W_in[0], a_in[0], oW_in[0], oa_in[0],
                  repo_users, repo_users, M_REPO_N, /*src_add=*/0, /*dst_const=*/U_N,
                  /*selfloop=*/true);
    }
    // ---------------- phase 2: user (self-loops inside edge list) --------------
    {
        const int n = U_N, npad = ((n + 127) / 128) * 128;
        (void)hipMemsetAsync(XB128, 0, (size_t)NMAX * 128 * sizeof(bf16), stream);
        cvt(XB128, YOUT, (size_t)U_N * 128);  // repo_h = rows [0, U)
        gat_block(n, npad, W_in[1], a_in[1], oW_in[1], oa_in[1],
                  user_edges, user_edges + E_USER_T, E_USER_T, 0, -1, /*selfloop=*/false);
    }
    // ---------------- phase 3: team (nodes = users + teams) --------------------
    {
        const int n = U_N + T_N, npad = ((n + 127) / 128) * 128;
        (void)hipMemsetAsync(XB128, 0, (size_t)NMAX * 128 * sizeof(bf16), stream);
        cvt(XB128, YOUT, (size_t)U_N * 128);                    // user_h
        cvt(XB128 + (size_t)U_N * 128, teams, (size_t)T_N * 128);
        gat_block(n, npad, W_in[2], a_in[2], oW_in[2], oa_in[2],
                  tu_team, tu_user, E_TEAM_N, /*src_add=*/U_N, /*dst_const=*/-1,
                  /*selfloop=*/true);
    }
    // ---------------- head: sigmoid(team_h @ out_W^T + b) ----------------------
    head_kernel<<<dim3((T_N + 255) / 256), 256, 0, stream>>>(
        out, YOUT + (size_t)U_N * 128, out_W, out_b, T_N);
}
